// RavenAttention_13846974562392
// MI455X (gfx1250) — matrix operand order repaired and checked by reading, not compiled, not run on hardware
//
#include <hip/hip_runtime.h>

typedef __attribute__((ext_vector_type(16))) __bf16 v16bf;
typedef __attribute__((ext_vector_type(8)))  __bf16 v8bf;
typedef __attribute__((ext_vector_type(8)))  float  v8f;
typedef int v4i32 __attribute__((vector_size(16)));   // matches builtin param

#define NHEAD   8
#define HD      128
#define TOPK    32
#define EPSV    1e-5f
#define LDS_PAD 40   // 32 data + 8 pad bf16 per row; 80B rows keep 16B alignment

#ifndef __has_builtin
#define __has_builtin(x) 0
#endif
#if __has_builtin(__builtin_amdgcn_global_load_async_to_lds_b128)
#define HAS_ASYNC_LDS 1
#else
#define HAS_ASYNC_LDS 0
#endif

__device__ __forceinline__ void wait_async_all() {
#if HAS_ASYNC_LDS
#if __has_builtin(__builtin_amdgcn_s_wait_asynccnt)
  __builtin_amdgcn_s_wait_asynccnt(0);
#else
  asm volatile("s_wait_asynccnt 0x0" ::: "memory");
#endif
#endif
}

__device__ __forceinline__ __bf16 f2bf(float f) {
  unsigned u = __builtin_bit_cast(unsigned, f);
  u += 0x7FFFu + ((u >> 16) & 1u);              // round-to-nearest-even
  unsigned short h = (unsigned short)(u >> 16);
  return __builtin_bit_cast(__bf16, h);
}

__device__ __forceinline__ float siluf(float x) { return x / (1.f + expf(-x)); }

// ---------------------------------------------------------------------------
// Elementwise fp32 -> bf16 (weights/activations pre-convert; L2-resident).
// ---------------------------------------------------------------------------
__global__ __launch_bounds__(256)
void cvt_f32_bf16(const float* __restrict__ src, __bf16* __restrict__ dst) {
  const size_t idx = ((size_t)blockIdx.x * 256 + threadIdx.x) * 8;
  const float4* s = (const float4*)(src + idx);
  float4 a = s[0], b = s[1];
  v8bf o;
  o[0] = f2bf(a.x); o[1] = f2bf(a.y); o[2] = f2bf(a.z); o[3] = f2bf(a.w);
  o[4] = f2bf(b.x); o[5] = f2bf(b.y); o[6] = f2bf(b.z); o[7] = f2bf(b.w);
  *(v8bf*)(dst + idx) = o;
}

// ---------------------------------------------------------------------------
// C[M x N] = A[M x K] * B[K x N]; A,B bf16 row-major, C f32.
// 256 threads / 128x128 C tile; wave grid 2x4, each wave 4x2 WMMA subtiles.
// Double-buffered LDS; A staged via GLOBAL_LOAD_ASYNC_TO_LDS_B128 when the
// builtin exists, B staged transposed (Bt[n][k]) for contiguous fragments.
// ---------------------------------------------------------------------------
__global__ __launch_bounds__(256)
void gemm_bf16_wmma(const __bf16* __restrict__ A, int lda,
                    const __bf16* __restrict__ B, int ldb,
                    float* __restrict__ C, int ldc, int Ksz) {
  __shared__ __bf16 As[2][128 * LDS_PAD];   // As[buf][m][k]
  __shared__ __bf16 Bt[2][128 * LDS_PAD];   // Bt[buf][n][k]
  const int tid  = threadIdx.x;
  const int lane = tid & 31;
  const int wave = tid >> 5;
  const int half = lane >> 4;
  const int lr   = lane & 15;
  const int bm0  = blockIdx.y * 128;
  const int bn0  = blockIdx.x * 128;
  const int wm0  = (wave >> 2) * 64;
  const int wn0  = (wave & 3) * 32;

  v8f acc[4][2];
  const v8f vz = {0.f,0.f,0.f,0.f,0.f,0.f,0.f,0.f};
#pragma unroll
  for (int i = 0; i < 4; i++)
#pragma unroll
    for (int j = 0; j < 2; j++) acc[i][j] = vz;

  auto stageA = [&](int kb, int nbuf) {
#pragma unroll
    for (int c = 0; c < 2; ++c) {                  // 512 8-elt chunks / 256 thr
      const int chunk = tid + c * 256;
      const int row = chunk >> 2;
      const int ko  = (chunk & 3) * 8;
      const __bf16* g = A + (size_t)(bm0 + row) * lda + kb + ko;
      __bf16* l = &As[nbuf][row * LDS_PAD + ko];
#if HAS_ASYNC_LDS
      __builtin_amdgcn_global_load_async_to_lds_b128(
          (__attribute__((address_space(1))) v4i32*)g,
          (__attribute__((address_space(3))) v4i32*)l, 0, 0);
#else
      *(v8bf*)l = *(const v8bf*)g;
#endif
    }
  };
  auto stageB = [&](int kb, int nbuf) {
    const int brow = tid >> 3, bc0 = (tid & 7) * 16;
    const v8bf* src = (const v8bf*)(B + (size_t)(kb + brow) * ldb + bn0 + bc0);
    v8bf b0 = src[0], b1 = src[1];
    __bf16* base = Bt[nbuf];
#pragma unroll
    for (int e = 0; e < 8; ++e) base[(bc0 + e) * LDS_PAD + brow] = b0[e];
#pragma unroll
    for (int e = 0; e < 8; ++e) base[(bc0 + 8 + e) * LDS_PAD + brow] = b1[e];
  };

  const int nk = Ksz >> 5;
  stageA(0, 0);
  stageB(0, 0);
  wait_async_all();
  __syncthreads();

  for (int it = 0; it < nk; ++it) {
    const int buf = it & 1;
    if (it + 1 < nk) {                 // prefetch next tile into other buffer
      stageA((it + 1) << 5, buf ^ 1);
      stageB((it + 1) << 5, buf ^ 1);
    }

    // ISA 16-bit A 16x32 layout: lane<16 -> row m, K {0..7,16..23};
    // lane>=16 -> same row, K {8..15,24..31}. Two 16B ds loads each.
    v16bf afrag[4], bfrag[2];
#pragma unroll
    for (int sm = 0; sm < 4; sm++) {
      const __bf16* ap = &As[buf][(wm0 + sm * 16 + lr) * LDS_PAD];
      v8bf lo = *(const v8bf*)(ap + half * 8);
      v8bf hi = *(const v8bf*)(ap + 16 + half * 8);
#pragma unroll
      for (int e = 0; e < 8; e++) { afrag[sm][e] = lo[e]; afrag[sm][e + 8] = hi[e]; }
    }
  // B 32x16 layout: lane -> column n0+lr, K {half*8..+7, 16+half*8..+7} (matches A)
#pragma unroll
    for (int sn = 0; sn < 2; sn++) {
      const __bf16* bp = &Bt[buf][(wn0 + sn * 16 + lr) * LDS_PAD];
      v8bf lo = *(const v8bf*)(bp + half * 8);
      v8bf hi = *(const v8bf*)(bp + 16 + half * 8);
#pragma unroll
      for (int e = 0; e < 8; e++) { bfrag[sn][e] = lo[e]; bfrag[sn][e + 8] = hi[e]; }
    }
#pragma unroll
    for (int sm = 0; sm < 4; sm++)
#pragma unroll
      for (int sn = 0; sn < 2; sn++)
        acc[sm][sn] = __builtin_amdgcn_wmma_f32_16x16x32_bf16(
            false, afrag[sm], false, bfrag[sn], (short)0, acc[sm][sn], false, false);

    wait_async_all();
    __syncthreads();
  }

  // D layout: VGPR v -> row m0 + half*8 + v, col n0 + lr
#pragma unroll
  for (int sm = 0; sm < 4; sm++)
#pragma unroll
    for (int sn = 0; sn < 2; sn++) {
      const int row0 = bm0 + wm0 + sm * 16 + half * 8;
      const int col  = bn0 + wn0 + sn * 16 + lr;
#pragma unroll
      for (int v = 0; v < 8; v++)
        C[(size_t)(row0 + v) * ldc + col] = acc[sm][sn][v];
    }
}

// ---------------------------------------------------------------------------
// Per-token epilogue: Wa GEMV + f_head, silu+RMSNorm(q,k), silu(v),
// sigmoid router + rank-based top-32 -> per-slot decay/gate. In-place on Y.
// Y row layout: [q(1024) | k(1024) | v(1024) | router->decay(1024)]
// ---------------------------------------------------------------------------
__global__ __launch_bounds__(256)
void post_proj(const float* __restrict__ hs, float* __restrict__ Y,
               float* __restrict__ gateb,
               const float* __restrict__ Wa, const float* __restrict__ A_log,
               const float* __restrict__ dt_bias, const float* __restrict__ qw,
               const float* __restrict__ kw) {
  const int tok = blockIdx.x;
  const int tid = threadIdx.x;
  __shared__ float pmat[256][NHEAD];
  __shared__ float fh[NHEAD];
  __shared__ float sred[256];
  __shared__ float ssc[256];

  float p[NHEAD];
#pragma unroll
  for (int h = 0; h < NHEAD; h++) p[h] = 0.f;
  for (int d = tid; d < 1024; d += 256) {
    float x = hs[(size_t)tok * 1024 + d];
    const float* wa = Wa + (size_t)d * NHEAD;
#pragma unroll
    for (int h = 0; h < NHEAD; h++) p[h] += x * wa[h];
  }
#pragma unroll
  for (int h = 0; h < NHEAD; h++) pmat[tid][h] = p[h];
  __syncthreads();
  if (tid < NHEAD) {
    float s = 0.f;
    for (int r = 0; r < 256; r++) s += pmat[r][tid];
    float x = s + dt_bias[tid];
    float sp = (x > 20.f) ? x : log1pf(expf(x));
    fh[tid] = -expf(A_log[tid]) * sp;
  }
  __syncthreads();

  const int sub = tid >> 7;         // 2 heads per pass
  const int i = tid & 127;
  const int seg = sub << 7;

  for (int hb = 0; hb < NHEAD; hb += 2) {
    const int h = hb + sub;
    const size_t rb = (size_t)tok * 4096 + h * HD;
    { // q: silu + RMSNorm
      float x = Y[rb + i];
      float sx = siluf(x);
      sred[tid] = sx * sx; __syncthreads();
      for (int off = 64; off >= 1; off >>= 1) { if (i < off) sred[tid] += sred[tid + off]; __syncthreads(); }
      float inv = rsqrtf(sred[seg] / 128.f + EPSV);
      __syncthreads();
      Y[rb + i] = sx * inv * qw[i];
    }
    { // k: silu + RMSNorm
      float x = Y[rb + 1024 + i];
      float sx = siluf(x);
      sred[tid] = sx * sx; __syncthreads();
      for (int off = 64; off >= 1; off >>= 1) { if (i < off) sred[tid] += sred[tid + off]; __syncthreads(); }
      float inv = rsqrtf(sred[seg] / 128.f + EPSV);
      __syncthreads();
      Y[rb + 1024 + i] = sx * inv * kw[i];
    }
    Y[rb + 2048 + i] = siluf(Y[rb + 2048 + i]);     // v: silu
    { // router: sigmoid, stable top-32 by rank, normalize, decay/gate
      float sc = 1.f / (1.f + expf(-Y[rb + 3072 + i]));
      ssc[tid] = sc; __syncthreads();
      int rank = 0;
      for (int j = 0; j < 128; j++) {
        float sj = ssc[seg + j];
        rank += (sj > sc) || (sj == sc && j < i);
      }
      float wv = (rank < TOPK) ? sc : 0.f;
      sred[tid] = wv; __syncthreads();
      for (int off = 64; off >= 1; off >>= 1) { if (i < off) sred[tid] += sred[tid + off]; __syncthreads(); }
      float wsum = sred[seg];
      __syncthreads();
      float wn  = wv / (wsum + 1e-9f);
      float dec = expf(fh[h] * wn);            // exp(f); ==1 where unselected
      Y[rb + 3072 + i] = dec;
      gateb[(size_t)tok * 1024 + h * HD + i] = 1.f - dec;
    }
    __syncthreads();
  }
}

// ---------------------------------------------------------------------------
// Gated linear-attention scan. Scan is independent per value-column j, so
// grid = (b,h) x 4 value-slices = 64 blocks. Each block: state 128x32 in
// registers (4x4 per thread, 32x8 thread grid). Partial o reduced in-wave
// with shfl_xor (lanes ^8,^16 share j), then LDS float atomics across waves.
// ---------------------------------------------------------------------------
__global__ __launch_bounds__(256)
void recurrence(const float* __restrict__ Y, const float* __restrict__ gateb,
                float* __restrict__ o_tmp) {
  const int blk = blockIdx.x;
  const int bh = blk >> 2;
  const int b = bh >> 3, h = bh & 7;
  const int jslice = (blk & 3) * 32;
  const int tid = threadIdx.x;
  const int i0 = (tid >> 3) * 4;     // 32 i-groups of 4 rows
  const int j0 = (tid & 7) * 4;      // 8 j-groups of 4 cols (within slice)
  __shared__ float qv[128], kv[128], dv[128], gv[128];
  __shared__ float vv[32], osm[32];
  float S[4][4];
#pragma unroll
  for (int a = 0; a < 4; a++)
#pragma unroll
    for (int c = 0; c < 4; c++) S[a][c] = 0.f;

  for (int t = 0; t < 1024; t++) {
    const int tok = b * 1024 + t;
    if (tid < 128) {
      const size_t base = (size_t)tok * 4096 + h * 128 + tid;
      qv[tid] = Y[base];
      kv[tid] = Y[base + 1024];
      dv[tid] = Y[base + 3072];
      gv[tid] = gateb[(size_t)tok * 1024 + h * 128 + tid];
    }
    if (tid < 32) {
      vv[tid] = Y[(size_t)tok * 4096 + h * 128 + 2048 + jslice + tid];
      osm[tid] = 0.f;
    }
    __syncthreads();
    float op[4];
#pragma unroll
    for (int c = 0; c < 4; c++) op[c] = 0.f;
#pragma unroll
    for (int a = 0; a < 4; a++) {
      const int i = i0 + a;
      const float d = dv[i];
      const float u = gv[i] * kv[i];
      const float qi = qv[i];
#pragma unroll
      for (int c = 0; c < 4; c++) {
        float s = d * S[a][c] + u * vv[j0 + c];
        S[a][c] = s;
        op[c] += qi * s;
      }
    }
#pragma unroll
    for (int c = 0; c < 4; c++) {          // in-wave: sum the 4 i-groups/wave
      op[c] += __shfl_xor(op[c], 8);
      op[c] += __shfl_xor(op[c], 16);
    }
    if ((tid & 24) == 0)
#pragma unroll
      for (int c = 0; c < 4; c++) atomicAdd(&osm[j0 + c], op[c]);
    __syncthreads();
    if (tid < 32)
      o_tmp[(size_t)tok * 1024 + h * 128 + jslice + tid] = osm[tid];
    __syncthreads();
  }
}

// ---------------------------------------------------------------------------
// Final RMSNorm over hidden (1024) with g_norm_w; writes bf16 for the Wo GEMM.
// ---------------------------------------------------------------------------
__global__ __launch_bounds__(256)
void gnorm_kernel(const float* __restrict__ o_tmp, const float* __restrict__ gw,
                  __bf16* __restrict__ O16) {
  const int tok = blockIdx.x;
  const int tid = threadIdx.x;
  __shared__ float sred[256];
  float x[4]; float ss = 0.f;
#pragma unroll
  for (int e = 0; e < 4; e++) {
    x[e] = o_tmp[(size_t)tok * 1024 + tid + e * 256];
    ss += x[e] * x[e];
  }
  sred[tid] = ss; __syncthreads();
  for (int off = 128; off >= 1; off >>= 1) { if (tid < off) sred[tid] += sred[tid + off]; __syncthreads(); }
  const float inv = rsqrtf(sred[0] / 1024.f + EPSV);
#pragma unroll
  for (int e = 0; e < 4; e++) {
    const int j = tid + e * 256;
    O16[(size_t)tok * 1024 + j] = f2bf(x[e] * inv * gw[j]);
  }
}

extern "C" void kernel_launch(void* const* d_in, const int* in_sizes, int n_in,
                              void* d_out, int out_size, void* d_ws, size_t ws_size,
                              hipStream_t stream) {
  (void)in_sizes; (void)n_in; (void)out_size; (void)ws_size;
  const float* hs      = (const float*)d_in[0];
  const float* Wq      = (const float*)d_in[1];
  const float* Wk      = (const float*)d_in[2];
  const float* Wv      = (const float*)d_in[3];
  const float* Wr      = (const float*)d_in[4];
  const float* Wa      = (const float*)d_in[5];
  const float* A_log   = (const float*)d_in[6];
  const float* dt_bias = (const float*)d_in[7];
  const float* qw      = (const float*)d_in[8];
  const float* kw      = (const float*)d_in[9];
  const float* gw      = (const float*)d_in[10];
  const float* Wo      = (const float*)d_in[11];
  float* out = (float*)d_out;

  // workspace layout
  float*  Y     = (float*)d_ws;                       // 2048*4096 f32 (32MB)
  float*  gateb = Y + (size_t)2048 * 4096;            // 2048*1024 f32 (8MB)
  float*  o_tmp = gateb + (size_t)2048 * 1024;        // 2048*1024 f32 (8MB)
  __bf16* X16   = (__bf16*)(o_tmp + (size_t)2048 * 1024);  // 2048*1024 bf16
  __bf16* Wq16  = X16 + (size_t)2048 * 1024;
  __bf16* Wk16  = Wq16 + (size_t)1024 * 1024;
  __bf16* Wv16  = Wk16 + (size_t)1024 * 1024;
  __bf16* Wr16  = Wv16 + (size_t)1024 * 1024;
  __bf16* Wo16  = Wr16 + (size_t)1024 * 1024;
  __bf16* O16   = Wo16 + (size_t)1024 * 1024;         // 2048*1024 bf16

  dim3 blk(256);
  dim3 gproj(1024 / 128, 2048 / 128);                 // (8, 16) tiles

  // pre-convert activations/weights to bf16 (L2-resident, trivially BW-bound)
  cvt_f32_bf16<<<1024, blk, 0, stream>>>(hs, X16);
  cvt_f32_bf16<<< 512, blk, 0, stream>>>(Wq, Wq16);
  cvt_f32_bf16<<< 512, blk, 0, stream>>>(Wk, Wk16);
  cvt_f32_bf16<<< 512, blk, 0, stream>>>(Wv, Wv16);
  cvt_f32_bf16<<< 512, blk, 0, stream>>>(Wr, Wr16);
  cvt_f32_bf16<<< 512, blk, 0, stream>>>(Wo, Wo16);

  gemm_bf16_wmma<<<gproj, blk, 0, stream>>>(X16, 1024, Wq16, 1024, Y + 0,    4096, 1024);
  gemm_bf16_wmma<<<gproj, blk, 0, stream>>>(X16, 1024, Wk16, 1024, Y + 1024, 4096, 1024);
  gemm_bf16_wmma<<<gproj, blk, 0, stream>>>(X16, 1024, Wv16, 1024, Y + 2048, 4096, 1024);
  gemm_bf16_wmma<<<gproj, blk, 0, stream>>>(X16, 1024, Wr16, 1024, Y + 3072, 4096, 1024);
  post_proj<<<2048, blk, 0, stream>>>(hs, Y, gateb, Wa, A_log, dt_bias, qw, kw);
  recurrence<<<64, blk, 0, stream>>>(Y, gateb, o_tmp);
  gnorm_kernel<<<2048, blk, 0, stream>>>(o_tmp, gw, O16);
  gemm_bf16_wmma<<<gproj, blk, 0, stream>>>(O16, 1024, Wo16, 1024, out, 1024, 1024);
}
